// QKVNormMultiHeadAttention_23897198035427
// MI455X (gfx1250) — matrix operand order repaired and checked by reading, not compiled, not run on hardware
//
#include <hip/hip_runtime.h>
#include <hip/hip_bf16.h>

// ---------------------------------------------------------------------------
// QKV-per-head-LayerNorm multi-head attention for MI455X (gfx1250, wave32).
// bf16 WMMA (v_wmma_f32_16x16x32_bf16) for all matmuls, f32 accumulate,
// flash-attention so (B,H,S,S) scores never touch HBM.
// CDNA5 data movement: TDM tensor_load_to_lds for weight tiles (TENSORcnt),
// global_load_async_to_lds_b128 for bf16 activation tiles (ASYNCcnt).
// Workspace (~72MB): bf16 W^T (8MB), Q/K bf16 [B][H][S][HD] (32MB),
// V bf16 transposed [B][H][HD][S] (16MB), attn-out bf16 [B][S][D] (16MB).
// ---------------------------------------------------------------------------

#define BATCH 4
#define SEQ   2048
#define DMODEL 1024
#define NHEAD 16
#define HDIM  64
#define MROWS (BATCH * SEQ) // 8192

typedef __attribute__((ext_vector_type(16))) __bf16 v16bf;
typedef __attribute__((ext_vector_type(8)))  float  v8f;
typedef __attribute__((ext_vector_type(4)))  unsigned int u32x4;
typedef __attribute__((ext_vector_type(8)))  int i32x8;
typedef __attribute__((ext_vector_type(4)))  int i32x4;

#if __has_builtin(__builtin_amdgcn_tensor_load_to_lds)
#define HAVE_TDM 1
#else
#define HAVE_TDM 0
#endif
#if __has_builtin(__builtin_amdgcn_global_load_async_to_lds_b128)
#define HAVE_ASYNC_LDS 1
#else
#define HAVE_ASYNC_LDS 0
#endif

union FragBF {
    u32x4 q[2];
    v16bf v;
};

__device__ __forceinline__ unsigned short f32_to_bf16(float f) {
    unsigned int u = __float_as_uint(f);
    u += 0x7FFFu + ((u >> 16) & 1u);   // round to nearest even
    return (unsigned short)(u >> 16);
}

__device__ __forceinline__ unsigned int pack2_bf16(float a, float b) {
    return (unsigned int)f32_to_bf16(a) | ((unsigned int)f32_to_bf16(b) << 16);
}

__device__ __forceinline__ v8f wmma_bf16(v16bf a, v16bf b, v8f c) {
    return __builtin_amdgcn_wmma_f32_16x16x32_bf16(false, a, false, b,
                                                   (short)0, c, false, false);
}

// A-fragment (16x32 bf16): lane holds row (lane&15); elements 0..7 -> K =
// khalf..khalf+7, elements 8..15 -> K = khalf+16..+23, khalf = 8*(lane>=16).
__device__ __forceinline__ FragBF load_a_frag(const unsigned short* row_ptr, int lane) {
    FragBF f;
    int khalf = (lane >> 4) << 3;
    f.q[0] = *(const u32x4*)(row_ptr + khalf);
    f.q[1] = *(const u32x4*)(row_ptr + khalf + 16);
    return f;
}

// B-fragment (32x16 bf16): lane holds column (lane&15); elements 0..7 -> K =
// kbase..kbase+7, elements 8..15 -> K = kbase+16..+23, kbase = 8*(lane>=16).
__device__ __forceinline__ FragBF load_b_frag(const unsigned short* col_ptr, int lane) {
    FragBF f;
    int kbase = (lane >> 4) << 3;
    f.q[0] = *(const u32x4*)(col_ptr + kbase);
    f.q[1] = *(const u32x4*)(col_ptr + kbase + 16);
    return f;
}

// ---- LDS byte offset of a __shared__ object (for TDM descriptors) ----
__device__ __forceinline__ unsigned lds_offset_of(const void* p) {
    return (unsigned)(size_t)(__attribute__((address_space(3))) const void*)p;
}

// ---- 16B global -> LDS stage: async DMA if available, else plain copy ----
__device__ __forceinline__ void stage16_g2l(void* lptr, const void* gptr) {
#if HAVE_ASYNC_LDS
    __builtin_amdgcn_global_load_async_to_lds_b128(
        (__attribute__((address_space(1))) i32x4*)(
            (__attribute__((address_space(1))) void*)(void*)gptr),
        (__attribute__((address_space(3))) i32x4*)(
            (__attribute__((address_space(3))) void*)lptr),
        0, 0);
#else
    *(u32x4*)lptr = *(const u32x4*)gptr;
#endif
}

#if HAVE_TDM
// Issue one TDM 2D tile load: bf16 tensor [tensor_d1][tensor_d0] (row stride
// stride0 elements), tile [tile_d1][tile_d0] starting at gptr -> LDS lds_off,
// packed contiguously. D# bit layout per CDNA5 ISA ch.8 (data_size=1 -> 2B).
__device__ __forceinline__ void tdm_load_2d(unsigned lds_off, const void* gptr,
                                            int tensor_d0, int tensor_d1,
                                            int tile_d0, int tile_d1, int stride0) {
    unsigned long long ga = (unsigned long long)(size_t)gptr;
    u32x4 g0;
    g0[0] = 1u;                                   // count=1 (valid user D#)
    g0[1] = lds_off;                              // lds_addr
    g0[2] = (unsigned)(ga & 0xFFFFFFFFu);         // global_addr[31:0]
    g0[3] = (unsigned)((ga >> 32) & 0x01FFFFFFu)  // global_addr[56:32]
            | (2u << 30);                         // type=2 ("image")
    i32x8 g1;
    g1[0] = (int)(1u << 16);                      // data_size=1 (2B), wg_mask=0
    g1[1] = (int)((unsigned)tensor_d0 << 16);     // tensor_dim0[15:0]
    g1[2] = (int)(((unsigned)tensor_d0 >> 16) |   // tensor_dim0[31:16]
                  ((unsigned)tensor_d1 << 16));   // tensor_dim1[15:0]
    g1[3] = (int)(((unsigned)tensor_d1 >> 16) |   // tensor_dim1[31:16]
                  ((unsigned)tile_d0 << 16));     // tile_dim0
    g1[4] = (int)((unsigned)tile_d1 & 0xFFFFu);   // tile_dim1 (tile_dim2=0)
    g1[5] = stride0;                              // tensor_dim0_stride[31:0]
    g1[6] = 0;
    g1[7] = 0;
    i32x4 z4 = {0, 0, 0, 0};
#if defined(__clang_major__) && __clang_major__ >= 23
    i32x8 z8 = {0, 0, 0, 0, 0, 0, 0, 0};
    __builtin_amdgcn_tensor_load_to_lds(g0, g1, z4, z4, z8, 0);
#else
    __builtin_amdgcn_tensor_load_to_lds(g0, g1, z4, z4, 0);
#endif
}
#endif

// ---------------------------------------------------------------------------
// Weight cast+transpose: W f32 [K][N] -> W^T bf16 [N][K]  (N = K = 1024)
// ---------------------------------------------------------------------------
__global__ __launch_bounds__(1024) void cast_transpose_kernel(
    const float* __restrict__ in, unsigned short* __restrict__ out) {
    __shared__ float tile[32][33];
    int x0 = blockIdx.x * 32, y0 = blockIdx.y * 32;
    int tx = threadIdx.x, ty = threadIdx.y;
    tile[ty][tx] = in[(size_t)(y0 + ty) * DMODEL + x0 + tx];
    __syncthreads();
    out[(size_t)(x0 + ty) * DMODEL + y0 + tx] = f32_to_bf16(tile[tx][ty]);
}

// ---------------------------------------------------------------------------
// GEMM (+bias) with optional fused per-head LayerNorm.
//   C[m, n0+0..63] = A[m, :] @ W[:, n0+0..63] + bias   (Wt holds W^T)
// mode 0: LN, bf16 out [B][H][S][HD]   (Q)
// mode 1: LN, bf16 out [B][H][S][HD]   (K)
// mode 2: LN, bf16 out [B][H][HD][S]   (V transposed)
// mode 3: no LN, f32 out [M][N]        (output projection, A is bf16)
// Block: 256 threads (8 waves). Tile 128x64. Wave tile 32x32.
// ---------------------------------------------------------------------------
struct SmemLoad {
    unsigned short a[128 * 32];   // A tile  [m][k]
    unsigned short bt[64 * 32];   // B tile  [n][k]   (from W^T)
};
union SmemGemm {
    SmemLoad load;
    float ctile[128 * 64];
};

__global__ __launch_bounds__(256) void gemm_ln_kernel(
    const void* __restrict__ Ain,            // f32 (mode<3) or bf16 (mode 3)
    const unsigned short* __restrict__ Wt,   // bf16 W^T [N=1024][K=1024]
    const float* __restrict__ bias,          // [1024]
    const float* __restrict__ ln_scale,      // [64] (modes 0-2)
    const float* __restrict__ ln_bias,       // [64]
    void* __restrict__ out,
    int mode)
{
    __shared__ SmemGemm sm;

    const int tid  = threadIdx.x;
    const int lane = tid & 31;
    const int wave = tid >> 5;
    const int wm = (wave & 3) * 32;
    const int wn = (wave >> 2) * 32;
    const int lrow = lane & 15;
    const int hi8  = (lane >> 4) << 3;

    const int m0 = blockIdx.x * 128;
    const int n0 = blockIdx.y * 64;

    const float* Af = (const float*)Ain;
    const unsigned short* Ab = (const unsigned short*)Ain;

    v8f acc[2][2];
    const v8f vzero = {0.f, 0.f, 0.f, 0.f, 0.f, 0.f, 0.f, 0.f};
    for (int i = 0; i < 2; ++i)
        for (int j = 0; j < 2; ++j)
            acc[i][j] = vzero;

    for (int kt = 0; kt < DMODEL / 32; ++kt) {
        const int k0 = kt * 32;

        // ---- stage B tile [64 n][32 k] from W^T: one TDM DMA per k-step ----
#if HAVE_TDM
        if (wave == 0)
            tdm_load_2d(lds_offset_of(&sm.load.bt[0]),
                        Wt + (size_t)n0 * DMODEL + k0,
                        DMODEL, DMODEL, /*tile_d0=*/32, /*tile_d1=*/64, DMODEL);
#else
        if (tid < 128) {
            int n = tid >> 2, kc = (tid & 3) << 3;
            stage16_g2l(&sm.load.bt[n * 32 + kc],
                        Wt + (size_t)(n0 + n) * DMODEL + k0 + kc);
        }
#endif

        // ---- stage A tile (128x32) ----
        for (int e = 0; e < 2; ++e) {
            int c    = tid + 256 * e;      // 0..511 chunks of 8 elements
            int mrow = c >> 2;
            int kc   = (c & 3) << 3;
            if (mode < 3) {
                // f32 source: convert to bf16 in VALU while staging
                const float* p = Af + (size_t)(m0 + mrow) * DMODEL + k0 + kc;
                float4 v0 = *(const float4*)(p);
                float4 v1 = *(const float4*)(p + 4);
                u32x4 w;
                w[0] = pack2_bf16(v0.x, v0.y);
                w[1] = pack2_bf16(v0.z, v0.w);
                w[2] = pack2_bf16(v1.x, v1.y);
                w[3] = pack2_bf16(v1.z, v1.w);
                *(u32x4*)(&sm.load.a[mrow * 32 + kc]) = w;
            } else {
                // bf16 source: async DMA straight into LDS
                stage16_g2l(&sm.load.a[mrow * 32 + kc],
                            Ab + (size_t)(m0 + mrow) * DMODEL + k0 + kc);
            }
        }
        if (kt + 1 < DMODEL / 32 && mode < 3)
            __builtin_prefetch(Af + (size_t)(m0 + (tid >> 1)) * DMODEL + k0 + 32, 0, 1);

#if HAVE_TDM
        __builtin_amdgcn_s_wait_tensorcnt(0);
#endif
#if HAVE_ASYNC_LDS
        __builtin_amdgcn_s_wait_asynccnt(0);
#endif
        __syncthreads();

        FragBF afr[2], bfr[2];
        for (int i = 0; i < 2; ++i)
            afr[i] = load_a_frag(&sm.load.a[(wm + i * 16 + lrow) * 32], lane);
        for (int j = 0; j < 2; ++j)
            bfr[j] = load_b_frag(&sm.load.bt[(wn + j * 16 + lrow) * 32], lane);

        for (int i = 0; i < 2; ++i)
            for (int j = 0; j < 2; ++j)
                acc[i][j] = wmma_bf16(afr[i].v, bfr[j].v, acc[i][j]);

        __syncthreads();
    }

    // ---- epilogue: fragments + bias -> LDS f32 tile ----
    for (int i = 0; i < 2; ++i)
        for (int j = 0; j < 2; ++j)
            for (int r = 0; r < 8; ++r) {
                int mr = wm + i * 16 + r + hi8;
                int nc = wn + j * 16 + lrow;
                sm.ctile[mr * 64 + nc] = acc[i][j][r] + bias[n0 + nc];
            }
    __syncthreads();

    if (mode == 3) {
        float* o = (float*)out;
        for (int e = 0; e < 32; ++e) {
            int c = tid + 256 * e;
            int row = c >> 6, col = c & 63;
            o[(size_t)(m0 + row) * DMODEL + n0 + col] = sm.ctile[row * 64 + col];
        }
        return;
    }

    // ---- per-head LayerNorm over this head's 64 columns ----
    if (tid < 128) {
        int row = tid;
        int m = m0 + row;
        int b = m >> 11;          // m / SEQ
        int s = m & (SEQ - 1);
        int h = n0 >> 6;
        const float* ct = &sm.ctile[row * 64];
        float mu = 0.f;
        for (int c = 0; c < 64; ++c) mu += ct[c];
        mu *= (1.0f / 64.0f);
        float var = 0.f;
        for (int c = 0; c < 64; ++c) { float d = ct[c] - mu; var += d * d; }
        var *= (1.0f / 64.0f);
        float rstd = rsqrtf(var + 1e-6f);
        unsigned short* ob = (unsigned short*)out;
        for (int c = 0; c < 64; ++c) {
            float y = (ct[c] - mu) * rstd * ln_scale[c] + ln_bias[c];
            size_t idx;
            if (mode == 2)   // V transposed: [B][H][HD][S]
                idx = ((size_t)((b * NHEAD + h) * HDIM + c)) * SEQ + s;
            else             // Q / K: [B][H][S][HD]
                idx = ((size_t)((b * NHEAD + h) * SEQ + s)) * HDIM + c;
            ob[idx] = f32_to_bf16(y);
        }
    }
}

// ---------------------------------------------------------------------------
// Flash attention over one (b, h, 64-row q-tile). 256 threads / 8 waves;
// each wave owns 2 of 16 16x16 fragments. Online softmax in LDS.
// ---------------------------------------------------------------------------
__global__ __launch_bounds__(256) void flash_attn_kernel(
    const unsigned short* __restrict__ Q,   // [B][H][S][HD] bf16
    const unsigned short* __restrict__ K,   // [B][H][S][HD] bf16
    const unsigned short* __restrict__ Vt,  // [B][H][HD][S] bf16
    unsigned short* __restrict__ AO)        // [B][S][D] bf16
{
    __shared__ float          sS[64 * 64];
    __shared__ unsigned short sP[64 * 64];
    __shared__ float sM[64], sL[64], sAlpha[64];

    const int q0 = blockIdx.x * 64;
    const int h  = blockIdx.y;
    const int b  = blockIdx.z;

    const size_t head = ((size_t)(b * NHEAD + h)) * SEQ * HDIM;
    const unsigned short* Qh = Q + head;
    const unsigned short* Kh = K + head;
    const unsigned short* Vh = Vt + head;   // [HD][S]

    const int tid  = threadIdx.x;
    const int lane = tid & 31;
    const int wave = tid >> 5;
    const int lrow = lane & 15;
    const int hi8  = (lane >> 4) << 3;
    const int f0   = wave * 2;

    const v8f vzero = {0.f, 0.f, 0.f, 0.f, 0.f, 0.f, 0.f, 0.f};
    v8f oacc[2];
    oacc[0] = vzero;
    oacc[1] = vzero;

    if (tid < 64) { sM[tid] = -1e30f; sL[tid] = 0.f; }
    __syncthreads();

    for (int j0 = 0; j0 < SEQ; j0 += 64) {
        // ---- scores: S = (Q K^T) / sqrt(HD) ----
        for (int fi = 0; fi < 2; ++fi) {
            int f = f0 + fi, mi = f >> 2, ni = f & 3;
            v8f acc = vzero;
            for (int kb = 0; kb < 2; ++kb) {
                FragBF a = load_a_frag(Qh + (size_t)(q0 + mi * 16 + lrow) * HDIM + kb * 32, lane);
                FragBF bb = load_b_frag(Kh + (size_t)(j0 + ni * 16 + lrow) * HDIM + kb * 32, lane);
                acc = wmma_bf16(a.v, bb.v, acc);
            }
            for (int r = 0; r < 8; ++r)
                sS[(mi * 16 + r + hi8) * 64 + ni * 16 + lrow] = acc[r] * 0.125f;
        }
        __syncthreads();

        // ---- online softmax stats (one thread per q-row) ----
        if (tid < 64) {
            int row = tid;
            float mo = sM[row], mx = mo;
            for (int c = 0; c < 64; ++c) mx = fmaxf(mx, sS[row * 64 + c]);
            float alpha = __expf(mo - mx);
            float sum = 0.f;
            for (int c = 0; c < 64; ++c) {
                float p = __expf(sS[row * 64 + c] - mx);
                sum += p;
                sP[row * 64 + c] = f32_to_bf16(p);
            }
            sM[row] = mx;
            sL[row] = sL[row] * alpha + sum;
            sAlpha[row] = alpha;
        }
        __syncthreads();

        // ---- O = O*alpha + P @ V ----
        for (int fi = 0; fi < 2; ++fi) {
            int f = f0 + fi, mi = f >> 2, ni = f & 3;
            for (int r = 0; r < 8; ++r)
                oacc[fi][r] *= sAlpha[mi * 16 + r + hi8];
            for (int kb = 0; kb < 2; ++kb) {
                FragBF a = load_a_frag(&sP[(mi * 16 + lrow) * 64 + kb * 32], lane);
                FragBF bb = load_b_frag(Vh + (size_t)(ni * 16 + lrow) * SEQ + j0 + kb * 32, lane);
                oacc[fi] = wmma_bf16(a.v, bb.v, oacc[fi]);
            }
        }
        __syncthreads();
    }

    // ---- epilogue: O / l -> bf16 attn-out [B][S][D] ----
    for (int fi = 0; fi < 2; ++fi) {
        int f = f0 + fi, mi = f >> 2, ni = f & 3;
        for (int r = 0; r < 8; ++r) {
            int row = mi * 16 + r + hi8;
            int hd  = ni * 16 + lrow;
            float val = oacc[fi][r] / sL[row];
            AO[((size_t)b * SEQ + (q0 + row)) * DMODEL + h * HDIM + hd] = f32_to_bf16(val);
        }
    }
}

// ---------------------------------------------------------------------------
// Launch
// ---------------------------------------------------------------------------
extern "C" void kernel_launch(void* const* d_in, const int* in_sizes, int n_in,
                              void* d_out, int out_size, void* d_ws, size_t ws_size,
                              hipStream_t stream) {
    const float* Xq  = (const float*)d_in[0];
    const float* Xkv = (const float*)d_in[1];
    const float* Wq  = (const float*)d_in[2];
    const float* bq  = (const float*)d_in[3];
    const float* Wk  = (const float*)d_in[4];
    const float* bk  = (const float*)d_in[5];
    const float* Wv  = (const float*)d_in[6];
    const float* bv  = (const float*)d_in[7];
    const float* qs  = (const float*)d_in[8];
    const float* qb  = (const float*)d_in[9];
    const float* ks  = (const float*)d_in[10];
    const float* kb  = (const float*)d_in[11];
    const float* vs  = (const float*)d_in[12];
    const float* vb  = (const float*)d_in[13];
    const float* Wo  = (const float*)d_in[14];
    const float* bo  = (const float*)d_in[15];

    char* ws = (char*)d_ws;
    size_t off = 0;
    auto alloc = [&](size_t bytes) {
        size_t o = off;
        off = (off + bytes + 255) & ~(size_t)255;
        return o;
    };
    const size_t wbytes = (size_t)DMODEL * DMODEL * 2;              // 2 MB
    const size_t qkvbytes = (size_t)BATCH * NHEAD * SEQ * HDIM * 2; // 16 MB
    unsigned short* Wq_t = (unsigned short*)(ws + alloc(wbytes));
    unsigned short* Wk_t = (unsigned short*)(ws + alloc(wbytes));
    unsigned short* Wv_t = (unsigned short*)(ws + alloc(wbytes));
    unsigned short* Wo_t = (unsigned short*)(ws + alloc(wbytes));
    unsigned short* Qb   = (unsigned short*)(ws + alloc(qkvbytes));
    unsigned short* Kb   = (unsigned short*)(ws + alloc(qkvbytes));
    unsigned short* Vtb  = (unsigned short*)(ws + alloc(qkvbytes));
    unsigned short* AOb  = (unsigned short*)(ws + alloc(qkvbytes));
    // total ~72 MB of workspace

    dim3 tgrid(DMODEL / 32, DMODEL / 32);   // (32, 32)
    dim3 tblk(32, 32);
    cast_transpose_kernel<<<tgrid, tblk, 0, stream>>>(Wq, Wq_t);
    cast_transpose_kernel<<<tgrid, tblk, 0, stream>>>(Wk, Wk_t);
    cast_transpose_kernel<<<tgrid, tblk, 0, stream>>>(Wv, Wv_t);
    cast_transpose_kernel<<<tgrid, tblk, 0, stream>>>(Wo, Wo_t);

    dim3 ggrid(MROWS / 128, DMODEL / 64);   // (64, 16)
    gemm_ln_kernel<<<ggrid, 256, 0, stream>>>(Xq,  Wq_t, bq, qs, qb, Qb,  0);
    gemm_ln_kernel<<<ggrid, 256, 0, stream>>>(Xkv, Wk_t, bk, ks, kb, Kb,  1);
    gemm_ln_kernel<<<ggrid, 256, 0, stream>>>(Xkv, Wv_t, bv, vs, vb, Vtb, 2);

    dim3 fgrid(SEQ / 64, NHEAD, BATCH);     // (32, 16, 4)
    flash_attn_kernel<<<fgrid, 256, 0, stream>>>(Qb, Kb, Vtb, AOb);

    gemm_ln_kernel<<<ggrid, 256, 0, stream>>>(AOb, Wo_t, bo, nullptr, nullptr, d_out, 3);
}
